// SGNN_26173530702436
// MI455X (gfx1250) — matrix operand order repaired and checked by reading, not compiled, not run on hardware
//
#include <hip/hip_runtime.h>
#include <math.h>

// ---------------------------------------------------------------------------
// SGNN forward for MI455X (gfx1250, wave32).
// Heavy GEMMs use V_WMMA_F32_16X16X32_BF16. edge->fc_edge path is refactored
// algebraically (rank-1 structure of the edge tensor) into a fused WMMA GEMM
// whose left operand relu(A_i + rs_i * s_k) is generated in registers; its
// LDS staging uses the Tensor Data Mover (tensor_load_to_lds + TENSORcnt).
// Workspace use ~70 MB (all L2-resident on 192MB L2).
// ---------------------------------------------------------------------------

typedef __bf16 bf16_t;
typedef bf16_t v16bf __attribute__((ext_vector_type(16)));
typedef bf16_t v8bf  __attribute__((ext_vector_type(8)));
typedef float  v8f   __attribute__((ext_vector_type(8)));
typedef unsigned int u32x4 __attribute__((ext_vector_type(4)));
typedef int          i32x8 __attribute__((ext_vector_type(8)));
typedef int          i32x4 __attribute__((ext_vector_type(4)));

#define Bn 2
#define Nn 768
#define Tt 12
#define Cc 32
#define NHh 8

#ifndef __has_builtin
#define __has_builtin(x) 0
#endif
#if __has_builtin(__builtin_amdgcn_tensor_load_to_lds) && \
    __has_builtin(__builtin_amdgcn_s_wait_tensorcnt)
#define USE_TDM 1
#else
#define USE_TDM 0
#endif

__device__ __forceinline__ v16bf cat8(v8bf lo, v8bf hi) {
  return __builtin_shufflevector(lo, hi, 0,1,2,3,4,5,6,7,8,9,10,11,12,13,14,15);
}
__device__ __forceinline__ float sigm(float x) { return 1.0f / (1.0f + __expf(-x)); }

#if USE_TDM
// TDM: async-load a contiguous run of `nelem` f32 from global into LDS.
// D# per CDNA5 ISA ch.8: group0 = {count=1, lds_addr, global_addr, type=2},
// group1 = {data_size=4B, tensor_dim0=tile_dim0=nelem, tile_dim1=1,
//           tensor_dim0_stride=nelem}. Groups 2/3 zero (<=2D tensor).
// This toolchain exposes the 6-arg builtin (extra int32x8 group, then cpol).
__device__ __forceinline__ void tdm_load_f32(unsigned lds_off, const void* gptr,
                                             int nelem) {
  unsigned long long ga = (unsigned long long)gptr;
  u32x4 g0;
  g0[0] = 1u;                                           // count=1 (valid), user
  g0[1] = lds_off;                                      // lds_addr (bytes)
  g0[2] = (unsigned)(ga & 0xffffffffu);                 // global_addr[31:0]
  g0[3] = (unsigned)((ga >> 32) & 0x01ffffffu)          // global_addr[56:32]
        | 0x80000000u;                                  // type=2 ("image")
  i32x8 g1;
  g1[0] = 0x00020000;                                   // data_size=2 -> 4B
  g1[1] = (nelem & 0xffff) << 16;                       // tensor_dim0[15:0]
  g1[2] = (1 << 16) | ((nelem >> 16) & 0xffff);         // tensor_dim1=1 | td0 hi
  g1[3] = (nelem & 0xffff) << 16;                       // tile_dim0
  g1[4] = 1;                                            // tile_dim1=1, tile_dim2=0
  g1[5] = nelem;                                        // tensor_dim0_stride lo
  g1[6] = 0;
  g1[7] = 0;
  i32x4 z4 = {0, 0, 0, 0};
  i32x8 z8 = {0, 0, 0, 0, 0, 0, 0, 0};
  __builtin_amdgcn_tensor_load_to_lds(g0, g1, z4, z4, z8, 0);
}
#endif

// ---------------------------------------------------------------------------
// Generic bf16 WMMA GEMM:  D[m,n] = sum_k A[m,k] * Bt[n,k]
// A row-major [M x K] bf16 (lda elems), Bt row-major [N x K] bf16 (ldb elems),
// D f32 (ldd elems). grid = (M/64, N/16, batch), block = 128 (4 waves).
// K must be a multiple of 32.  A/B fragment layouts per CDNA5 ISA 7.12.2.
// ---------------------------------------------------------------------------
__global__ __launch_bounds__(128) void gemm_bf16_wmma(
    const bf16_t* __restrict__ A, size_t sA, int lda,
    const bf16_t* __restrict__ Bt, size_t sB, int ldb,
    float* __restrict__ D, size_t sD, int ldd, int K)
{
  const int bidx = blockIdx.z;
  A  += (size_t)bidx * sA;
  Bt += (size_t)bidx * sB;
  D  += (size_t)bidx * sD;
  const int lane = threadIdx.x & 31;
  const int wave = threadIdx.x >> 5;
  const int hi   = lane >> 4;       // lane group (0: lanes 0-15, 1: lanes 16-31)
  const int l16  = lane & 15;
  const int m0 = blockIdx.x * 64 + wave * 16;
  const int n0 = blockIdx.y * 16;
  // A: lane holds row m0+l16; K elems {8*hi..8*hi+7} and {16+8*hi..16+8*hi+7}
  const bf16_t* pa = A + (size_t)(m0 + l16) * lda + 8 * hi;
  // B: lane holds col n0+l16; K elems {8*hi..8*hi+7} and {16+8*hi..16+8*hi+7}
  const bf16_t* pb = Bt + (size_t)(n0 + l16) * ldb + 8 * hi;
  v8f acc = {};
  for (int kk = 0; kk < K; kk += 32) {
    v8bf alo = *(const v8bf*)(pa + kk);
    v8bf ahi = *(const v8bf*)(pa + kk + 16);
    v8bf blo = *(const v8bf*)(pb + kk);
    v8bf bhi = *(const v8bf*)(pb + kk + 16);
    __builtin_prefetch(pa + kk + 32, 0, 0);   // global_prefetch_b8
    __builtin_prefetch(pb + kk + 32, 0, 0);
    acc = __builtin_amdgcn_wmma_f32_16x16x32_bf16(
        false, cat8(alo, ahi), false, cat8(blo, bhi),
        (short)0, acc, false, false);
  }
  // D layout: lane col = n0+l16, VGPR r -> row m0 + 8*hi + r
  const int rbase = m0 + 8 * hi;
  #pragma unroll
  for (int r = 0; r < 8; ++r)
    D[(size_t)(rbase + r) * ldd + (n0 + l16)] = acc[r];
}

// ---------------------------------------------------------------------------
// adj -> bf16 copy + row sums.  grid=768, block=256.
// ---------------------------------------------------------------------------
__global__ void k_adj_pack(const float* __restrict__ adj,
                           bf16_t* __restrict__ adj_bf,
                           float* __restrict__ rs)
{
  const int i = blockIdx.x, tid = threadIdx.x;
  __shared__ float red[256];
  float s = 0.f;
  for (int j = tid; j < Nn; j += 256) {
    float a = adj[i * Nn + j];
    adj_bf[i * Nn + j] = (bf16_t)a;
    s += a;
  }
  red[tid] = s; __syncthreads();
  for (int o = 128; o > 0; o >>= 1) {
    if (tid < o) red[tid] += red[tid + o];
    __syncthreads();
  }
  if (tid == 0) rs[i] = red[0];
}

// ---------------------------------------------------------------------------
// Pack small weights: wqkv_bt [224][96], fce_bt [32][768], colsum[32].
// grid=1, block=256.
// ---------------------------------------------------------------------------
__global__ void k_wpack(const float* __restrict__ wq, const float* __restrict__ wk,
                        const float* __restrict__ wv, const float* __restrict__ fcew,
                        bf16_t* __restrict__ wqkv_bt, bf16_t* __restrict__ fce_bt,
                        float* __restrict__ colsum)
{
  const int tid = threadIdx.x;
  for (int idx = tid; idx < 224 * 96; idx += 256) {
    int n = idx / 96, k = idx % 96;
    float w = (n < 64)  ? wq[k * 64 + n]
            : (n < 128) ? wk[k * 64 + (n - 64)]
                        : wv[k * 96 + (n - 128)];
    wqkv_bt[n * 96 + k] = (bf16_t)w;
  }
  for (int idx = tid; idx < 32 * Nn; idx += 256) {
    int o = idx / Nn, i = idx % Nn;
    fce_bt[o * Nn + i] = (bf16_t)fcew[i * 32 + o];
  }
  if (tid < 32) {
    float s = 0.f;
    for (int i = 0; i < Nn; ++i) s += fcew[i * 32 + tid];
    colsum[tid] = s;
  }
}

// ---------------------------------------------------------------------------
// Prep per (b,n): em, x0 = x@enc_w+enc_b; write bf16-transposed RHS for adj
// GEMM (channels 0..95 = x0, 96..107 = em, 108..111 = 0).  grid=1536, blk=128.
// ---------------------------------------------------------------------------
__global__ void k_prep(const float* __restrict__ x,
                       const float* __restrict__ enc_w, const float* __restrict__ enc_b,
                       float* __restrict__ em_f, bf16_t* __restrict__ rhs1_bt,
                       float* __restrict__ x0_f, bf16_t* __restrict__ x0_rm)
{
  const int b = blockIdx.x / Nn, n = blockIdx.x % Nn, tid = threadIdx.x;
  const size_t bn = (size_t)b * Nn + n;
  __shared__ float xs[Tt * Cc];
  for (int idx = tid; idx < Tt * Cc; idx += 128)
    xs[idx] = x[bn * Tt * Cc + idx];
  __syncthreads();
  if (tid < 16) {
    if (tid < Tt) {
      float s = 0.f;
      for (int c = 0; c < Cc; ++c) s += xs[tid * Cc + c];
      s *= (1.0f / Cc);
      em_f[bn * Tt + tid] = s;
      rhs1_bt[((size_t)b * 112 + 96 + tid) * Nn + n] = (bf16_t)s;
    } else {
      rhs1_bt[((size_t)b * 112 + 96 + tid) * Nn + n] = (bf16_t)0.0f; // pad 108..111
    }
  }
  if (tid < 96) {
    int t = tid >> 3, m = tid & 7;
    float s = enc_b[m];
    for (int c = 0; c < Cc; ++c) s += xs[t * Cc + c] * enc_w[c * 8 + m];
    x0_f[bn * 96 + tid] = s;
    x0_rm[bn * 96 + tid] = (bf16_t)s;
    rhs1_bt[((size_t)b * 112 + tid) * Nn + n] = (bf16_t)s;
  }
}

// ---------------------------------------------------------------------------
// Node ODE 1 + edge coefficient prep.  C1 = [adj@x0 | adj@em] (b,n,112).
// x0b = x0 + relu(ax@theta_node); AmatT/BvT stored [b][t][n] (transposed,
// contiguous per (b,t) for the TDM load in k_xe_wmma).  grid=1536, block=96.
// ---------------------------------------------------------------------------
__global__ void k_node_edge(const float* __restrict__ C1, const float* __restrict__ x0_f,
                            const float* __restrict__ em_f,
                            const float* __restrict__ thn, const float* __restrict__ the,
                            float* __restrict__ x0b, float* __restrict__ AmatT,
                            float* __restrict__ BvT)
{
  const int b = blockIdx.x / Nn, n = blockIdx.x % Nn, tid = threadIdx.x;
  const size_t bn = (size_t)b * Nn + n;
  __shared__ float axs[96], ams[12], ems[12];
  axs[tid] = C1[bn * 112 + tid];
  if (tid < 12) { ams[tid] = C1[bn * 112 + 96 + tid]; ems[tid] = em_f[bn * Tt + tid]; }
  __syncthreads();
  {
    int t = tid >> 3, m2 = tid & 7;
    float v = 0.f;
    for (int m = 0; m < 8; ++m) v += axs[t * 8 + m] * thn[m * 8 + m2];
    x0b[bn * 96 + tid] = x0_f[bn * 96 + tid] + fmaxf(v, 0.f); // TIME = 1.0
  }
  if (tid < 12) {
    float sa = 0.f, sb = 0.f;
    for (int t = 0; t < Tt; ++t) {
      float th = the[t * Tt + tid];
      sa += ams[t] * th;
      sb += ems[t] * th;
    }
    AmatT[((size_t)b * Tt + tid) * Nn + n] = sa;
    BvT[((size_t)b * Tt + tid) * Nn + n]   = sb;
  }
}

// ---------------------------------------------------------------------------
// QKV bias + pack: q_bf,k_bf [b][h][n][32] (cols 8..31 zero, q scaled),
// v_bt [b][h][16][n] (rows 12..15 zero).  grid=1536, block=256.
// ---------------------------------------------------------------------------
__global__ void k_qkvpack(const float* __restrict__ qkv,
                          const float* __restrict__ bq, const float* __restrict__ bk,
                          const float* __restrict__ bv,
                          bf16_t* __restrict__ q_bf, bf16_t* __restrict__ k_bf,
                          bf16_t* __restrict__ v_bt)
{
  const int b = blockIdx.x / Nn, n = blockIdx.x % Nn, tid = threadIdx.x;
  const float SCALE = 0.35355339059327373f; // 1/sqrt(8)
  __shared__ float qs[224];
  const size_t bn = (size_t)b * Nn + n;
  if (tid < 224) qs[tid] = qkv[bn * 224 + tid];
  __syncthreads();
  for (int idx = tid; idx < 640; idx += 256) {
    if (idx < 256) {
      int h = idx >> 5, col = idx & 31;
      float v = (col < 8) ? (qs[h * 8 + col] + bq[h * 8 + col]) * SCALE : 0.f;
      q_bf[(((size_t)b * NHh + h) * Nn + n) * 32 + col] = (bf16_t)v;
    } else if (idx < 512) {
      int j = idx - 256, h = j >> 5, col = j & 31;
      float v = (col < 8) ? (qs[64 + h * 8 + col] + bk[h * 8 + col]) : 0.f;
      k_bf[(((size_t)b * NHh + h) * Nn + n) * 32 + col] = (bf16_t)v;
    } else {
      int j = idx - 512, h = j >> 4, d = j & 15;
      float v = (d < 12) ? (qs[128 + h * 12 + d] + bv[h * 12 + d]) : 0.f;
      v_bt[(((size_t)b * NHh + h) * 16 + d) * Nn + n] = (bf16_t)v;
    }
  }
}

// ---------------------------------------------------------------------------
// Row softmax over 768, output bf16 probabilities.  grid=16*768, block=256.
// ---------------------------------------------------------------------------
__global__ void k_softmax(const float* __restrict__ S, bf16_t* __restrict__ P)
{
  const int tid = threadIdx.x;
  const float* row = S + (size_t)blockIdx.x * Nn;
  bf16_t* prow = P + (size_t)blockIdx.x * Nn;
  __shared__ float red[256];
  float a0 = row[tid], a1 = row[tid + 256], a2 = row[tid + 512];
  float mx = fmaxf(a0, fmaxf(a1, a2));
  red[tid] = mx; __syncthreads();
  for (int o = 128; o > 0; o >>= 1) { if (tid < o) red[tid] = fmaxf(red[tid], red[tid + o]); __syncthreads(); }
  mx = red[0]; __syncthreads();
  float e0 = __expf(a0 - mx), e1 = __expf(a1 - mx), e2 = __expf(a2 - mx);
  red[tid] = e0 + e1 + e2; __syncthreads();
  for (int o = 128; o > 0; o >>= 1) { if (tid < o) red[tid] += red[tid + o]; __syncthreads(); }
  float inv = 1.0f / red[0];
  prow[tid]       = (bf16_t)(e0 * inv);
  prow[tid + 256] = (bf16_t)(e1 * inv);
  prow[tid + 512] = (bf16_t)(e2 * inv);
}

// ---------------------------------------------------------------------------
// Assemble x1 (concat heads) + bf16 transpose for adj GEMM. grid=1536, blk=96.
// ---------------------------------------------------------------------------
__global__ void k_x1(const float* __restrict__ att, float* __restrict__ x1_f,
                     bf16_t* __restrict__ x1_bt)
{
  const int b = blockIdx.x / Nn, n = blockIdx.x % Nn, c = threadIdx.x;
  const int h = c / 12, d = c % 12;
  float v = att[(((size_t)b * NHh + h) * Nn + n) * 16 + d];
  x1_f[((size_t)b * Nn + n) * 96 + c] = v;
  x1_bt[((size_t)b * 96 + c) * Nn + n] = (bf16_t)v;
}

// ---------------------------------------------------------------------------
// Node ODE 2 (4 slices of 24): O = x1 + relu((adj@x1)@theta24). grid=1536, blk=96.
// ---------------------------------------------------------------------------
__global__ void k_ode2(const float* __restrict__ C2, const float* __restrict__ x1_f,
                       const float* __restrict__ th1, float* __restrict__ O)
{
  const int b = blockIdx.x / Nn, n = blockIdx.x % Nn, tid = threadIdx.x;
  const size_t bn = (size_t)b * Nn + n;
  __shared__ float c2s[96], x1s[96];
  c2s[tid] = C2[bn * 96 + tid];
  x1s[tid] = x1_f[bn * 96 + tid];
  __syncthreads();
  int i = tid / 24, mm = tid % 24;
  float v = 0.f;
  for (int m = 0; m < 24; ++m) v += c2s[i * 24 + m] * th1[m * 24 + mm];
  O[bn * 96 + tid] = x1s[tid] + fmaxf(v, 0.f);
}

// ---------------------------------------------------------------------------
// Pc[b][t][o] = sum_i fce_w[i,o] * em[b,i,t].  grid=24, block=256.
// ---------------------------------------------------------------------------
__global__ void k_pconst(const float* __restrict__ fcew, const float* __restrict__ em_f,
                         float* __restrict__ Pc)
{
  const int b = blockIdx.x / Tt, t = blockIdx.x % Tt, tid = threadIdx.x;
  const int o = tid & 31, ch = tid >> 5;
  __shared__ float red[256];
  float s = 0.f;
  for (int i = ch * 96; i < ch * 96 + 96; ++i)
    s += fcew[i * 32 + o] * em_f[((size_t)b * Nn + i) * Tt + t];
  red[tid] = s; __syncthreads();
  if (tid < 32) {
    float acc = 0.f;
    for (int c = 0; c < 8; ++c) acc += red[c * 32 + tid];
    Pc[((size_t)b * Tt + t) * 32 + tid] = acc;
  }
}

// ---------------------------------------------------------------------------
// Fused edge GEMM with on-the-fly left matrix:
//   XE[b,k,t,o] = sum_i relu(AmatT[b,t,i] + rs[i]*BvT[b,t,k]) * fce_w[i,o]
// A/rs vectors staged into LDS via Tensor Data Mover (async, TENSORcnt).
// grid = (12 m-tiles, 2 n-tiles, 24 = b*t), block = 128 (4 waves).
// ---------------------------------------------------------------------------
__global__ __launch_bounds__(128) void k_xe_wmma(
    const float* __restrict__ AmatT, const float* __restrict__ BvT,
    const float* __restrict__ rs, const bf16_t* __restrict__ fce_bt,
    float* __restrict__ XE)
{
  const int b = blockIdx.z / Tt, t = blockIdx.z % Tt;
  const int tid = threadIdx.x;
  __shared__ __align__(16) float sA[Nn];
  __shared__ __align__(16) float sR[Nn];
  const float* Arow = AmatT + (size_t)(b * Tt + t) * Nn;
#if USE_TDM
  if (tid < 32) {  // wave 0 issues both DMAs (TDM issues once per wave)
    tdm_load_f32((unsigned)(unsigned long long)(void*)sA, Arow, Nn);
    tdm_load_f32((unsigned)(unsigned long long)(void*)sR, rs, Nn);
    __builtin_amdgcn_s_wait_tensorcnt(0);
  }
  __syncthreads();
#else
  for (int i = tid; i < Nn; i += 128) {
    sA[i] = Arow[i];
    sR[i] = rs[i];
  }
  __syncthreads();
#endif
  const int lane = tid & 31, wave = tid >> 5;
  const int hi = lane >> 4, l16 = lane & 15;
  const int m0 = blockIdx.x * 64 + wave * 16;
  const int n0 = blockIdx.y * 16;
  const int mrow = m0 + l16;
  const float srow = BvT[(size_t)(b * Tt + t) * Nn + mrow];
  const bf16_t* pb = fce_bt + (size_t)(n0 + l16) * Nn + 8 * hi;
  v8f acc = {};
  for (int kk = 0; kk < Nn; kk += 32) {
    const int i0 = kk + 8 * hi;
    v16bf af;
    #pragma unroll
    for (int e = 0; e < 8; ++e) {
      af[e]     = (bf16_t)fmaxf(sA[i0 + e]      + sR[i0 + e]      * srow, 0.f);
      af[e + 8] = (bf16_t)fmaxf(sA[i0 + 16 + e] + sR[i0 + 16 + e] * srow, 0.f);
    }
    v8bf blo = *(const v8bf*)(pb + kk);
    v8bf bhi = *(const v8bf*)(pb + kk + 16);
    acc = __builtin_amdgcn_wmma_f32_16x16x32_bf16(
        false, af, false, cat8(blo, bhi), (short)0, acc, false, false);
  }
  const int rbase = m0 + 8 * hi;
  #pragma unroll
  for (int r = 0; r < 8; ++r)
    XE[(((size_t)b * Nn + (rbase + r)) * Tt + t) * 32 + (n0 + l16)] = acc[r];
}

// ---------------------------------------------------------------------------
// Final: torch-scramble gather of O -> x1s, clip, fc1/fc2/fc3 + edge consts,
// sigmoid gating, relu.  grid=1536, block=384 (thread = (t,o)).
// ---------------------------------------------------------------------------
__global__ __launch_bounds__(384) void k_final(
    const float* __restrict__ x, const float* __restrict__ x0b,
    const float* __restrict__ O, const float* __restrict__ em_f,
    const float* __restrict__ XE, const float* __restrict__ Pc,
    const float* __restrict__ colsum,
    const float* __restrict__ fc1w, const float* __restrict__ fc1b,
    const float* __restrict__ fc2w, const float* __restrict__ fc2b,
    const float* __restrict__ fc3w, const float* __restrict__ fc3b,
    const float* __restrict__ fceb,
    const float* __restrict__ clipp, const float* __restrict__ wts,
    float* __restrict__ out)
{
  const int b = blockIdx.x / Nn, n = blockIdx.x % Nn, tid = threadIdx.x;
  const size_t bn = (size_t)b * Nn + n;
  __shared__ float res_s[384], x0b_s[96], g_s[96];
  res_s[tid] = x[bn * 384 + tid];
  if (tid < 96) x0b_s[tid] = x0b[bn * 96 + tid];
  __syncthreads();
  const float clipv = clipp[0];
  if (tid < 96) {
    // x1s[b,n,t2,c2] = O[c2&1, (n%64)*12+t2, (c2>>1)*24 + (b*12 + n/64)]
    int t2 = tid >> 3, c2 = tid & 7;
    int bb = c2 & 1, i = c2 >> 1;
    int m = b * 12 + (n >> 6);
    int srcn = (n & 63) * 12 + t2;
    float val = O[((size_t)bb * Nn + srcn) * 96 + i * 24 + m];
    float xv = x0b_s[tid];
    float v1 = (xv + clipv - val < 0.f) ? (xv + clipv) : val;
    float v2 = ((xv - clipv) - v1 > 0.f) ? (xv - clipv) : v1;
    g_s[tid] = v2;
  }
  __syncthreads();
  const int t = tid >> 5, o = tid & 31;
  float xg = fc1b[o], xl = fc2b[o];
  #pragma unroll
  for (int m = 0; m < 8; ++m) {
    xg += g_s[t * 8 + m] * fc1w[m * 32 + o];
    xl += x0b_s[t * 8 + m] * fc2w[m * 32 + o];
  }
  float emv = em_f[bn * Tt + t];
  float xe = XE[(bn * Tt + t) * 32 + o] + Pc[((size_t)b * Tt + t) * 32 + o]
           + emv * colsum[o] + fceb[o];
  float sg = sigm(xg), sl = sigm(xl), se = sigm(xe);
  float outg = (xg * sl + xg * se + xl * se + xl * sg + xe * sl + xe * sg) * (1.0f / 6.0f);
  float z3 = fc3b[o];
  #pragma unroll
  for (int c = 0; c < 32; ++c) z3 += res_s[t * 32 + c] * fc3w[c * 32 + o];
  float z = wts[0] * sigm(z3) + wts[1] * outg;
  out[(bn * Tt + t) * 32 + o] = fmaxf(z, 0.f);
}

// ---------------------------------------------------------------------------
extern "C" void kernel_launch(void* const* d_in, const int* in_sizes, int n_in,
                              void* d_out, int out_size, void* d_ws, size_t ws_size,
                              hipStream_t stream)
{
  const float* x    = (const float*)d_in[0];
  const float* adj  = (const float*)d_in[1];
  const float* encw = (const float*)d_in[2];
  const float* encb = (const float*)d_in[3];
  const float* thn  = (const float*)d_in[4];
  const float* the  = (const float*)d_in[5];
  const float* th1  = (const float*)d_in[6];
  const float* wq   = (const float*)d_in[7];
  const float* bq   = (const float*)d_in[8];
  const float* wk   = (const float*)d_in[9];
  const float* bk   = (const float*)d_in[10];
  const float* wv   = (const float*)d_in[11];
  const float* bv   = (const float*)d_in[12];
  const float* fc1w = (const float*)d_in[13];
  const float* fc1b = (const float*)d_in[14];
  const float* fc2w = (const float*)d_in[15];
  const float* fc2b = (const float*)d_in[16];
  const float* fc3w = (const float*)d_in[17];
  const float* fc3b = (const float*)d_in[18];
  const float* fcew = (const float*)d_in[19];
  const float* fceb = (const float*)d_in[20];
  const float* clip = (const float*)d_in[21];
  const float* wts  = (const float*)d_in[22];
  float* out = (float*)d_out;

  // ---- carve workspace (256B aligned) ----
  size_t off = 0;
  char* base = (char*)d_ws;
  auto carve = [&](size_t bytes) -> void* {
    void* p = base + off;
    off += (bytes + 255) & ~(size_t)255;
    return p;
  };
  bf16_t* adj_bf  = (bf16_t*)carve((size_t)Nn * Nn * 2);
  float*  rsv     = (float*) carve(Nn * 4);
  bf16_t* rhs1_bt = (bf16_t*)carve((size_t)Bn * 112 * Nn * 2);
  float*  x0_f    = (float*) carve((size_t)Bn * Nn * 96 * 4);
  bf16_t* x0_rm   = (bf16_t*)carve((size_t)Bn * Nn * 96 * 2);
  float*  em_f    = (float*) carve((size_t)Bn * Nn * Tt * 4);
  float*  C1      = (float*) carve((size_t)Bn * Nn * 112 * 4);
  bf16_t* wqkv_bt = (bf16_t*)carve((size_t)224 * 96 * 2);
  bf16_t* fce_bt  = (bf16_t*)carve((size_t)32 * Nn * 2);
  float*  colsum  = (float*) carve(32 * 4);
  float*  qkv     = (float*) carve((size_t)Bn * Nn * 224 * 4);
  bf16_t* q_bf    = (bf16_t*)carve((size_t)Bn * NHh * Nn * 32 * 2);
  bf16_t* k_bf    = (bf16_t*)carve((size_t)Bn * NHh * Nn * 32 * 2);
  bf16_t* v_bt    = (bf16_t*)carve((size_t)Bn * NHh * 16 * Nn * 2);
  float*  S       = (float*) carve((size_t)Bn * NHh * Nn * Nn * 4);
  bf16_t* P_bf    = (bf16_t*)carve((size_t)Bn * NHh * Nn * Nn * 2);
  float*  att     = (float*) carve((size_t)Bn * NHh * Nn * 16 * 4);
  float*  x1_f    = (float*) carve((size_t)Bn * Nn * 96 * 4);
  bf16_t* x1_bt   = (bf16_t*)carve((size_t)Bn * 96 * Nn * 2);
  float*  C2      = (float*) carve((size_t)Bn * Nn * 96 * 4);
  float*  Obuf    = (float*) carve((size_t)Bn * Nn * 96 * 4);
  float*  x0b     = (float*) carve((size_t)Bn * Nn * 96 * 4);
  float*  AmatT   = (float*) carve((size_t)Bn * Tt * Nn * 4);
  float*  BvT     = (float*) carve((size_t)Bn * Tt * Nn * 4);
  float*  XE      = (float*) carve((size_t)Bn * Nn * Tt * 32 * 4);
  float*  Pc      = (float*) carve((size_t)Bn * Tt * 32 * 4);
  (void)ws_size; (void)n_in; (void)in_sizes; (void)out_size;

  const int BNgrid = Bn * Nn; // 1536

  // 1) adjacency pack + rowsums; weight packs; per-(b,n) prep
  k_adj_pack<<<Nn, 256, 0, stream>>>(adj, adj_bf, rsv);
  k_wpack<<<1, 256, 0, stream>>>(wq, wk, wv, fcew, wqkv_bt, fce_bt, colsum);
  k_prep<<<BNgrid, 128, 0, stream>>>(x, encw, encb, em_f, rhs1_bt, x0_f, x0_rm);

  // 2) C1 = adj @ [x0 | em | pad]  (M=768, N=112, K=768)
  gemm_bf16_wmma<<<dim3(12, 7, Bn), 128, 0, stream>>>(
      adj_bf, 0, Nn, rhs1_bt, (size_t)112 * Nn, Nn,
      C1, (size_t)Nn * 112, 112, Nn);

  // 3) node ODE 1 + edge coefficients
  k_node_edge<<<BNgrid, 96, 0, stream>>>(C1, x0_f, em_f, thn, the, x0b, AmatT, BvT);

  // 4) QKV = x0 @ [wq|wk|wv]  (M=768, N=224, K=96)
  gemm_bf16_wmma<<<dim3(12, 14, Bn), 128, 0, stream>>>(
      x0_rm, (size_t)Nn * 96, 96, wqkv_bt, 0, 96,
      qkv, (size_t)Nn * 224, 224, 96);
  k_qkvpack<<<BNgrid, 256, 0, stream>>>(qkv, bq, bk, bv, q_bf, k_bf, v_bt);

  // 5) scores = q @ k^T (scaled), per (b,h): M=768, N=768, K=32(8 real)
  gemm_bf16_wmma<<<dim3(12, 48, Bn * NHh), 128, 0, stream>>>(
      q_bf, (size_t)Nn * 32, 32, k_bf, (size_t)Nn * 32, 32,
      S, (size_t)Nn * Nn, Nn, 32);
  k_softmax<<<Bn * NHh * Nn, 256, 0, stream>>>(S, P_bf);

  // 6) att = P @ v, per (b,h): M=768, N=16(12 real), K=768
  gemm_bf16_wmma<<<dim3(12, 1, Bn * NHh), 128, 0, stream>>>(
      P_bf, (size_t)Nn * Nn, Nn, v_bt, (size_t)16 * Nn, Nn,
      att, (size_t)Nn * 16, 16, Nn);
  k_x1<<<BNgrid, 96, 0, stream>>>(att, x1_f, x1_bt);

  // 7) C2 = adj @ x1  (M=768, N=96, K=768), node ODE 2
  gemm_bf16_wmma<<<dim3(12, 6, Bn), 128, 0, stream>>>(
      adj_bf, 0, Nn, x1_bt, (size_t)96 * Nn, Nn,
      C2, (size_t)Nn * 96, 96, Nn);
  k_ode2<<<BNgrid, 96, 0, stream>>>(C2, x1_f, th1, Obuf);

  // 8) fc_edge: rank-1 consts + fused relu-generated WMMA GEMM (TDM staging)
  k_pconst<<<Bn * Tt, 256, 0, stream>>>(fcew, em_f, Pc);
  k_xe_wmma<<<dim3(12, 2, Bn * Tt), 128, 0, stream>>>(AmatT, BvT, rsv, fce_bt, XE);

  // 9) gather/clip/gating/final
  k_final<<<BNgrid, 384, 0, stream>>>(
      x, x0b, Obuf, em_f, XE, Pc, colsum,
      fc1w, fc1b, fc2w, fc2b, fc3w, fc3b, fceb, clip, wts, out);
}
